// MultiScaleDeformableAttention_88596585382323
// MI455X (gfx1250) — matrix operand order repaired and checked by reading, not compiled, not run on hardware
//
#include <hip/hip_runtime.h>

// Multi-Scale Deformable Attention for MI455X (gfx1250, wave32, WMMA).
// bs=2, nq=nv=21760, D=256, nh=8, dh=32, L=4, P=4.
// Pipeline:
//   prep:   f32->bf16 convert of query/value; weight transpose+convert to (N,K) bf16
//   GEMM1:  v    = value @ Wv + bv         -> bf16 (gather operand, halves L2 traffic)
//   GEMM2:  offs = query @ Wo + bo         -> f32
//   GEMM3:  attw = query @ Wa + ba         -> f32
//   sample: fused 16-wide softmax + bilinear gather, 1 wave per (b,q,head),
//           1 channel per lane (dh==32==wave32)     -> bf16
//   GEMM4:  out  = msda @ Wout + bout + query (residual) -> f32
// GEMMs: v_wmma_f32_16x16x32_bf16. Each wave computes a 16x64 strip: the A
// fragment is loaded once per k-step and reused across 4 accumulators. All 4 B
// fragments are loaded before the 4 WMMAs so waits retire progressively instead
// of draining per WMMA. A (M,K) bf16 and Bt (N,K) bf16 layouts make every
// fragment two contiguous b128 loads.

#define NH 8
#define NL 4
#define NP 4
#define DH 32
#define DMODEL 256

typedef __attribute__((ext_vector_type(16))) __bf16 v16bf;
typedef __attribute__((ext_vector_type(8)))  __bf16 v8bf;
typedef __attribute__((ext_vector_type(4)))  __bf16 v4bf;
typedef __attribute__((ext_vector_type(8)))  float  v8f;
typedef __attribute__((ext_vector_type(4)))  float  v4f;

__device__ __forceinline__ __bf16 f2bf(float f) {
  unsigned u = __builtin_bit_cast(unsigned, f);
  u += 0x7FFFu + ((u >> 16) & 1u);                 // round-to-nearest-even
  unsigned short s = (unsigned short)(u >> 16);
  return __builtin_bit_cast(__bf16, s);
}
__device__ __forceinline__ float bf2f(__bf16 x) {
  unsigned u = ((unsigned)__builtin_bit_cast(unsigned short, x)) << 16;
  return __builtin_bit_cast(float, u);
}

// ---- prep: elementwise f32 -> bf16 (vectorized x4) ----
__global__ __launch_bounds__(256) void cvt_f32_bf16(
    const float* __restrict__ src, __bf16* __restrict__ dst, int n)
{
  const int i = (blockIdx.x * blockDim.x + threadIdx.x) * 4;
  if (i + 3 >= n) {
    for (int j = i; j < n; ++j) dst[j] = f2bf(src[j]);
    return;
  }
  v4f x = *(const v4f*)(src + i);
  v4bf y;
#pragma unroll
  for (int j = 0; j < 4; ++j) y[j] = f2bf(x[j]);
  *(v4bf*)(dst + i) = y;
}

// ---- prep: W (K,N) f32 -> Wt (N,K) bf16 ----
__global__ __launch_bounds__(256) void transpose_cvt(
    const float* __restrict__ W, __bf16* __restrict__ Wt, int K, int N)
{
  const int idx = blockIdx.x * blockDim.x + threadIdx.x;   // output-linear
  if (idx >= K * N) return;
  const int n = idx / K, k = idx % K;
  Wt[idx] = f2bf(W[(size_t)k * N + n]);
}

// ---- WMMA GEMM: C[M,N] = A[M,K]*Bt[N,K]^T + bias (+residual) ----
// Wave computes a 16x64 strip (4 accumulators, shared A fragment).
// Block = 8 waves; waves tile (8/(N/64)) M-rows x (N/64) strips.
// grid = (M/16/rowsPerBlock, 1), block = 256. All strides compile-time.
template<int N, int K, bool RES, bool OUTBF>
__global__ __launch_bounds__(256) void gemm_wmma(
    const __bf16* __restrict__ A,        // (M,K) row-major
    const __bf16* __restrict__ Bt,       // (N,K) row-major (B transposed)
    const float*  __restrict__ bias,     // (N)
    const float*  __restrict__ residual, // (M,N) if RES
    float*        __restrict__ Cf,       // if !OUTBF
    __bf16*       __restrict__ Cb,       // if OUTBF
    int M)
{
  constexpr int STRIPS = N / 64;          // N-strips per 16-row band
  constexpr int ROWSPB = 8 / STRIPS;      // 16-row bands per block
  static_assert(8 % STRIPS == 0, "block must tile evenly");

  const int lane = threadIdx.x & 31;
  const int wave = threadIdx.x >> 5;
  const int mSub = wave / STRIPS;
  const int nSub = wave % STRIPS;
  const int tileM = blockIdx.x * ROWSPB + mSub;
  if (tileM * 16 >= M) return;            // wave-uniform

  const int hi   = lane >> 4;             // half-wave select
  const int rowA = tileM * 16 + (lane & 15);
  const int col0 = nSub * 64 + (lane & 15);

  // A lane: row rowA, K in {kb..kb+7} u {kb+16..kb+23}, kb = hi*8
  const __bf16* ap = A  + (size_t)rowA * K + hi * 8;
  // B lane: column col0 + 16*t, K in {kb..kb+7} u {kb+16..kb+23}, kb = hi*8
  const __bf16* bp = Bt + (size_t)col0 * K + hi * 8;

  v8f acc[4] = {{}, {}, {}, {}};
#pragma unroll
  for (int k0 = 0; k0 < K; k0 += 32) {
    // issue all 10 loads of this k-step up front, then 4 WMMAs retire against
    // progressively decreasing load counts
    v8bf a0 = *(const v8bf*)(ap + k0);          // b128
    v8bf a1 = *(const v8bf*)(ap + k0 + 16);     // b128
    v8bf b0[4], b1[4];
#pragma unroll
    for (int t = 0; t < 4; ++t) {
      const __bf16* bpt = bp + (size_t)(16 * t) * K + k0;
      b0[t] = *(const v8bf*)(bpt);              // b128
      b1[t] = *(const v8bf*)(bpt + 16);         // b128
    }
    v16bf af = __builtin_shufflevector(a0, a1, 0,1,2,3,4,5,6,7,8,9,10,11,12,13,14,15);
#pragma unroll
    for (int t = 0; t < 4; ++t) {
      v16bf bf = __builtin_shufflevector(b0[t], b1[t],
                                         0,1,2,3,4,5,6,7,8,9,10,11,12,13,14,15);
      acc[t] = __builtin_amdgcn_wmma_f32_16x16x32_bf16(
          false, af, false, bf, (short)0, acc[t], false, false);
    }
  }

  const int row0 = tileM * 16 + hi * 8;         // D elem r -> row row0+r
#pragma unroll
  for (int t = 0; t < 4; ++t) {
    const int col = col0 + 16 * t;
    const float bvv = bias[col];
#pragma unroll
    for (int r = 0; r < 8; ++r) {
      const size_t idx = (size_t)(row0 + r) * N + col;
      float val = acc[t][r] + bvv;
      if (RES)   val += residual[idx];
      if (OUTBF) Cb[idx] = f2bf(val);
      else       Cf[idx] = val;
    }
  }
}

// ---- fused softmax + bilinear sampling ----
// One wave per (b, q, head); lane = channel. v and out are bf16.
__global__ __launch_bounds__(256) void msda_sample(
    const __bf16* __restrict__ v,      // (bs, nv, nh, dh) bf16
    const float*  __restrict__ offs,   // (bs*nq, nh*L*P*2)
    const float*  __restrict__ logits, // (bs*nq, nh*L*P)
    const float*  __restrict__ ref,    // (bs, nq, L, 2)
    __bf16*       __restrict__ out,    // (bs, nq, nh*dh) bf16
    int bs, int nq, int nv)
{
  const int lane = threadIdx.x & 31;
  const int gw   = (blockIdx.x * blockDim.x + threadIdx.x) >> 5;
  const int total = bs * nq * NH;
  if (gw >= total) return;                          // wave-uniform

  const int h  = gw % NH;
  const int bq = gw / NH;                           // b*nq + q
  const int b  = bq / nq;

  // 16-wide softmax (redundant per lane; cheap, saves a global round-trip)
  const float* lg = logits + (size_t)bq * (NH * NL * NP) + h * (NL * NP);
  float lv[16];
  float mx = -3.0e38f;
#pragma unroll
  for (int i = 0; i < 16; ++i) { lv[i] = lg[i]; mx = fmaxf(mx, lv[i]); }
  float sum = 0.0f;
#pragma unroll
  for (int i = 0; i < 16; ++i) { lv[i] = __expf(lv[i] - mx); sum += lv[i]; }
  const float inv = 1.0f / sum;

  const float* of = offs + (size_t)bq * (NH * NL * NP * 2) + h * (NL * NP * 2);
  const float* rp = ref + (size_t)bq * (NL * 2);

  const int Hs[NL] = {128, 64, 32, 16};
  const int Ws[NL] = {128, 64, 32, 16};
  const int St[NL] = {0, 16384, 20480, 21504};

  float accv = 0.0f;
#pragma unroll
  for (int l = 0; l < NL; ++l) {
    const int Hl = Hs[l], Wl = Ws[l], s0 = St[l];
    const float rx = rp[l * 2 + 0], ry = rp[l * 2 + 1];
#pragma unroll
    for (int p = 0; p < NP; ++p) {
      const float ox = of[(l * NP + p) * 2 + 0];
      const float oy = of[(l * NP + p) * 2 + 1];
      // loc = ref + off/norm; x = loc*W - 0.5 (align_corners=False)
      const float x = (rx + ox / (float)Wl) * (float)Wl - 0.5f;
      const float y = (ry + oy / (float)Hl) * (float)Hl - 0.5f;
      const float xf = floorf(x), yf = floorf(y);
      const int   x0 = (int)xf,   y0 = (int)yf;
      const float wx1 = x - xf, wy1 = y - yf;
      const float wx0 = 1.0f - wx1, wy0 = 1.0f - wy1;
      const float w = lv[l * NP + p] * inv;

      float sample = 0.0f;
#pragma unroll
      for (int dy = 0; dy < 2; ++dy) {
        const int yi = y0 + dy;
        const bool vy = (yi >= 0) && (yi < Hl);
        const int yc = yi < 0 ? 0 : (yi > Hl - 1 ? Hl - 1 : yi);
        const float wy = dy ? wy1 : wy0;
#pragma unroll
        for (int dx = 0; dx < 2; ++dx) {
          const int xi = x0 + dx;
          const bool vx = (xi >= 0) && (xi < Wl);
          const int xc = xi < 0 ? 0 : (xi > Wl - 1 ? Wl - 1 : xi);
          const float wxv = dx ? wx1 : wx0;
          const size_t sidx = (size_t)b * nv + (size_t)(s0 + yc * Wl + xc);
          const float val = bf2f(v[(sidx * NH + h) * DH + lane]); // coalesced 64B
          sample += val * (wxv * wy * (float)(vx && vy));
        }
      }
      accv += w * sample;
    }
  }
  out[((size_t)bq * NH + h) * DH + lane] = f2bf(accv);
}

extern "C" void kernel_launch(void* const* d_in, const int* in_sizes, int n_in,
                              void* d_out, int out_size, void* d_ws, size_t ws_size,
                              hipStream_t stream) {
  const float* query = (const float*)d_in[0];
  const float* value = (const float*)d_in[1];
  const float* refpt = (const float*)d_in[2];
  // d_in[3] spatial_shapes, d_in[4] level_start_index: compile-time constants
  const float* Wv   = (const float*)d_in[5];
  const float* bv   = (const float*)d_in[6];
  const float* Wo   = (const float*)d_in[7];
  const float* bo   = (const float*)d_in[8];
  const float* Wa   = (const float*)d_in[9];
  const float* ba   = (const float*)d_in[10];
  const float* Wout = (const float*)d_in[11];
  const float* bout = (const float*)d_in[12];

  const int nq = 128 * 128 + 64 * 64 + 32 * 32 + 16 * 16;   // 21760
  const int nv = nq;
  const int bs = in_sizes[0] / (nq * DMODEL);               // 2
  const int M  = bs * nq;                                   // 43520
  const int NA = NH * NL * NP;                              // 128

  // ---- workspace carving (256B aligned) ----
  char*  ws  = (char*)d_ws;
  size_t off = 0;
  auto carve = [&](size_t bytes) {
    void* p = ws + off; off += (bytes + 255) & ~(size_t)255; return p;
  };
  __bf16* qbf      = (__bf16*)carve((size_t)M * DMODEL * 2);
  __bf16* vbf      = (__bf16*)carve((size_t)M * DMODEL * 2);
  __bf16* Wv_t     = (__bf16*)carve((size_t)DMODEL * DMODEL * 2);
  __bf16* Wo_t     = (__bf16*)carve((size_t)DMODEL * DMODEL * 2);
  __bf16* Wa_t     = (__bf16*)carve((size_t)NA * DMODEL * 2);
  __bf16* Wout_t   = (__bf16*)carve((size_t)DMODEL * DMODEL * 2);
  __bf16* v_buf    = (__bf16*)carve((size_t)M * DMODEL * 2);
  float*  offs_buf = (float*) carve((size_t)M * DMODEL * 4);
  float*  attw_buf = (float*) carve((size_t)M * NA * 4);
  __bf16* msda_buf = (__bf16*)carve((size_t)M * DMODEL * 2);

  dim3 blk(256);
  const int nElem = M * DMODEL;

  // ---- prep ----
  cvt_f32_bf16<<<dim3((nElem / 4 + 255) / 256), blk, 0, stream>>>(query, qbf, nElem);
  cvt_f32_bf16<<<dim3((nElem / 4 + 255) / 256), blk, 0, stream>>>(value, vbf, nElem);
  transpose_cvt<<<dim3((DMODEL * DMODEL + 255) / 256), blk, 0, stream>>>(Wv,   Wv_t,   DMODEL, DMODEL);
  transpose_cvt<<<dim3((DMODEL * DMODEL + 255) / 256), blk, 0, stream>>>(Wo,   Wo_t,   DMODEL, DMODEL);
  transpose_cvt<<<dim3((DMODEL * NA     + 255) / 256), blk, 0, stream>>>(Wa,   Wa_t,   DMODEL, NA);
  transpose_cvt<<<dim3((DMODEL * DMODEL + 255) / 256), blk, 0, stream>>>(Wout, Wout_t, DMODEL, DMODEL);

  // ---- GEMMs + sampling ----
  // N=256: 4 strips/row, 2 rows/block -> grid.x = M/16/2
  // N=128: 2 strips/row, 4 rows/block -> grid.x = M/16/4
  // 1) v = value @ Wv + bv  -> bf16
  gemm_wmma<DMODEL, DMODEL, false, true><<<dim3(M / 16 / 2), blk, 0, stream>>>(
      vbf, Wv_t, bv, nullptr, nullptr, v_buf, M);
  // 2) offs = query @ Wo + bo -> f32
  gemm_wmma<DMODEL, DMODEL, false, false><<<dim3(M / 16 / 2), blk, 0, stream>>>(
      qbf, Wo_t, bo, nullptr, offs_buf, nullptr, M);
  // 3) attw logits = query @ Wa + ba -> f32
  gemm_wmma<NA, DMODEL, false, false><<<dim3(M / 16 / 4), blk, 0, stream>>>(
      qbf, Wa_t, ba, nullptr, attw_buf, nullptr, M);
  // 4) fused softmax + bilinear sampling
  const int total_waves = M * NH;                 // 348160
  msda_sample<<<dim3((total_waves * 32 + 255) / 256), blk, 0, stream>>>(
      v_buf, offs_buf, attw_buf, refpt, msda_buf, bs, nq, nv);
  // 5) out = msda @ Wout + bout + query -> f32
  gemm_wmma<DMODEL, DMODEL, true, false><<<dim3(M / 16 / 2), blk, 0, stream>>>(
      msda_buf, Wout_t, bout, query, (float*)d_out, nullptr, M);
}
